// HgruRealV6_60997125538135
// MI455X (gfx1250) — matrix operand order repaired and checked by reading, not compiled, not run on hardware
//
#include <hip/hip_runtime.h>
#include <hip/hip_bf16.h>

// ---------------------------------------------------------------------------
// HGRU block on gfx1250 (MI455X): bf16 WMMA GEMMs + chunked scan + LN.
//   inp  = silu(x W_in^T + b_in)          (f32 out, feeds the scan exactly)
//   gate = silu(x W_gate^T + b_gate)      (bf16 out, only feeds a product)
//   lam  = sigmoid(x W_lam^T + b_lam)     (f32 out)
//   h_t  = lam_t h_{t-1} + (1-lam_t) inp_t   (3-pass chunked scan, in-place)
//   out  = (LN(h)*gate) W_out^T + b_out
// GEMM: wave tile 64x64 (16 WMMA tiles), 8 waves/block -> 128x256 block tile.
// ---------------------------------------------------------------------------

typedef __attribute__((ext_vector_type(16))) __bf16 v16bf;
typedef __attribute__((ext_vector_type(8)))  __bf16 v8bf;
typedef __attribute__((ext_vector_type(8)))  float  v8f;

#define SEQ_   2048
#define BATCH_ 8
#define DIM_   1024
#define MROWS_ (SEQ_ * BATCH_)          // 16384
#define NCH_   16                       // scan chunks
#define CSTEP_ (SEQ_ / NCH_)            // 128 steps per chunk
#define CHANS_ (BATCH_ * DIM_)          // 8192 independent recurrences

// ---------------------------------------------------------------- f32 -> bf16
__global__ void k_cvt_bf16(const float* __restrict__ in, __bf16* __restrict__ out, int n) {
  for (int i = blockIdx.x * blockDim.x + threadIdx.x; i < n; i += gridDim.x * blockDim.x)
    out[i] = (__bf16)in[i];
}

// ------------------------------------------------------- bf16 WMMA GEMM core
// C[M,N] = ACT( A[M,K] * W[N,K]^T + bias[N] )
// ACT: 0=none, 1=silu, 2=sigmoid.  OUTBF: 0 -> f32, 1 -> bf16.
template <int ACT, int OUTBF>
__global__ __launch_bounds__(256)
void k_gemm_wmma(const __bf16* __restrict__ A, const __bf16* __restrict__ W,
                 const float* __restrict__ bias, void* __restrict__ Cout,
                 int M, int N, int K)
{
  const int lane  = threadIdx.x & 31;
  const int wave  = threadIdx.x >> 5;
  const int waveM = wave & 1;                          // 2 wave rows
  const int waveN = wave >> 1;                         // 4 wave cols
  const int rowBase = blockIdx.y * 128 + waveM * 64;   // 4 row-tiles of 16
  const int colBase = blockIdx.x * 256 + waveN * 64;   // 4 col-tiles of 16
  const int mh   = lane & 15;                          // row/col within tile
  const int half = lane >> 4;                          // lane half -> K split

  v8f acc[4][4] = {};

  // A fragment as loaded here (contiguous K order, same as B): lanes 0-15 hold K {0..15},
  // lanes 16-31 hold K {16..31}, 2 values per VGPR in K order.
  const __bf16* Abase = A + (size_t)(rowBase + mh) * K;
  // B = W^T (K x N): lane n=mh holds K kh*16..kh*16+15 of column n, i.e. 16
  // contiguous bf16 of row n of W (W is row-major N x K).
  const __bf16* Wbase = W + (size_t)(colBase + mh) * K;

  for (int kt = 0; kt < K; kt += 32) {
    v16bf a[4];
#pragma unroll
    for (int r = 0; r < 4; ++r) {
      const __bf16* p = Abase + (size_t)r * 16 * K + kt + half * 16;
      v8bf lo = *(const v8bf*)(p);
      v8bf hi = *(const v8bf*)(p + 8);
      a[r] = __builtin_shufflevector(lo, hi, 0,1,2,3,4,5,6,7,8,9,10,11,12,13,14,15);
    }
    v16bf b[4];
#pragma unroll
    for (int c = 0; c < 4; ++c) {
      const __bf16* p = Wbase + (size_t)c * 16 * K + kt + half * 16;
      b[c] = *(const v16bf*)p;
      // Unconditional next-tile prefetch (speculative prefetch past the end
      // is architecturally dropped): keeps the weight stream ahead in L0/L2.
      __builtin_prefetch(p + 32, 0, 3);                // global_prefetch_b8
    }
#pragma unroll
    for (int r = 0; r < 4; ++r)
#pragma unroll
      for (int c = 0; c < 4; ++c)
        acc[r][c] = __builtin_amdgcn_wmma_f32_16x16x32_bf16(
            false, a[r], false, b[c], (short)0, acc[r][c], false, false);
  }

  // Epilogue. C/D layout: lanes 0-15 -> N=lane, M=v; lanes 16-31 -> N=lane-16, M=8+v.
#pragma unroll
  for (int r = 0; r < 4; ++r)
#pragma unroll
    for (int c = 0; c < 4; ++c) {
      const int n  = colBase + c * 16 + mh;
      const float bn = bias[n];
#pragma unroll
      for (int v = 0; v < 8; ++v) {
        const int m = rowBase + r * 16 + half * 8 + v;
        float s = acc[r][c][v] + bn;
        float y;
        if (ACT == 1)      y = s * (1.0f / (1.0f + __expf(-s)));   // silu
        else if (ACT == 2) y = 1.0f / (1.0f + __expf(-s));         // sigmoid
        else               y = s;
        const size_t idx = (size_t)m * N + n;
        if (OUTBF) ((__bf16*)Cout)[idx] = (__bf16)y;
        else       ((float*)Cout)[idx]  = y;
      }
    }
}

// ------------------------------------------------------------- scan: pass A
// Per (chunk, channel): A_c = prod(lam), U_c = local scan end with h_in = 0.
__global__ void k_scan_chunks(const float* __restrict__ lam, const float* __restrict__ inp,
                              float* __restrict__ Ac, float* __restrict__ Uc)
{
  const int ch = blockIdx.x * blockDim.x + threadIdx.x;      // 0..CHANS_-1
  const int c  = blockIdx.y;                                 // chunk
  if (ch >= CHANS_) return;
  size_t idx = (size_t)c * CSTEP_ * CHANS_ + ch;
  float A = 1.0f, U = 0.0f;
  for (int t = 0; t < CSTEP_; ++t, idx += CHANS_) {
    const float l = lam[idx];
    U = l * U + (1.0f - l) * inp[idx];
    A *= l;
  }
  Ac[(size_t)c * CHANS_ + ch] = A;
  Uc[(size_t)c * CHANS_ + ch] = U;
}

// ------------------------------------------------------------- scan: pass B
// Sequentially combine the NCH_ chunk summaries -> per-chunk carry-in.
__global__ void k_scan_carries(const float* __restrict__ Ac, const float* __restrict__ Uc,
                               float* __restrict__ carry)
{
  const int ch = blockIdx.x * blockDim.x + threadIdx.x;
  if (ch >= CHANS_) return;
  float h = 0.0f;
  for (int c = 0; c < NCH_; ++c) {
    carry[(size_t)c * CHANS_ + ch] = h;
    h = Ac[(size_t)c * CHANS_ + ch] * h + Uc[(size_t)c * CHANS_ + ch];
  }
}

// ------------------------------------------------------------- scan: pass C
// Replay each chunk from its carry; h overwrites inp in place.
__global__ void k_scan_apply(const float* __restrict__ lam, float* __restrict__ inp_h,
                             const float* __restrict__ carry)
{
  const int ch = blockIdx.x * blockDim.x + threadIdx.x;
  const int c  = blockIdx.y;
  if (ch >= CHANS_) return;
  float h = carry[(size_t)c * CHANS_ + ch];
  size_t idx = (size_t)c * CSTEP_ * CHANS_ + ch;
  for (int t = 0; t < CSTEP_; ++t, idx += CHANS_) {
    const float l = lam[idx];
    h = l * h + (1.0f - l) * inp_h[idx];
    inp_h[idx] = h;
  }
}

// ------------------------------------------- LayerNorm * gate -> bf16 A-operand
// One wave32 per row of 1024; lane handles 32 strided elements; shfl reduce.
__global__ __launch_bounds__(128)
void k_ln_gate(const float* __restrict__ h, const __bf16* __restrict__ gate,
               const float* __restrict__ gamma, const float* __restrict__ beta,
               __bf16* __restrict__ hg)
{
  const int lane = threadIdx.x & 31;
  const int row  = blockIdx.x * (blockDim.x >> 5) + (threadIdx.x >> 5);
  const float* hr = h + (size_t)row * DIM_;
  float vals[32];
  float sum = 0.0f, ssq = 0.0f;
#pragma unroll
  for (int i = 0; i < 32; ++i) {
    const float v = hr[lane + i * 32];
    vals[i] = v; sum += v; ssq += v * v;
  }
#pragma unroll
  for (int m = 16; m >= 1; m >>= 1) {
    sum += __shfl_xor(sum, m, 32);
    ssq += __shfl_xor(ssq, m, 32);
  }
  const float mu  = sum * (1.0f / DIM_);
  const float var = ssq * (1.0f / DIM_) - mu * mu;
  const float rs  = rsqrtf(var + 1e-5f);
#pragma unroll
  for (int i = 0; i < 32; ++i) {
    const int j = lane + i * 32;
    const float y = (vals[i] - mu) * rs * gamma[j] + beta[j];
    const float g = (float)gate[(size_t)row * DIM_ + j];
    hg[(size_t)row * DIM_ + j] = (__bf16)(y * g);
  }
}

// ---------------------------------------------------------------------------
extern "C" void kernel_launch(void* const* d_in, const int* in_sizes, int n_in,
                              void* d_out, int out_size, void* d_ws, size_t ws_size,
                              hipStream_t stream) {
  (void)in_sizes; (void)n_in; (void)out_size; (void)ws_size;

  const float* x      = (const float*)d_in[0];
  const float* W_in   = (const float*)d_in[1];
  const float* b_in   = (const float*)d_in[2];
  const float* W_lam  = (const float*)d_in[3];
  const float* b_lam  = (const float*)d_in[4];
  const float* W_gate = (const float*)d_in[5];
  const float* b_gate = (const float*)d_in[6];
  const float* W_out  = (const float*)d_in[7];
  const float* b_out  = (const float*)d_in[8];
  const float* gamma  = (const float*)d_in[9];
  const float* beta   = (const float*)d_in[10];

  const size_t MB = 1024 * 1024;
  char* ws = (char*)d_ws;
  __bf16* xb    = (__bf16*)(ws + 0);          // 32 MB bf16 x  (reused as hg)
  __bf16* wbin  = (__bf16*)(ws + 32 * MB);    // 2 MB each
  __bf16* wblam = (__bf16*)(ws + 34 * MB);
  __bf16* wbgat = (__bf16*)(ws + 36 * MB);
  __bf16* wbout = (__bf16*)(ws + 38 * MB);
  float*  inp   = (float*)(ws + 40 * MB);     // 64 MB f32 (becomes h in place)
  float*  lam   = (float*)(ws + 104 * MB);    // 64 MB f32
  __bf16* gateb = (__bf16*)(ws + 168 * MB);   // 32 MB bf16
  float*  Ac    = (float*)(ws + 200 * MB);    // 512 KB
  float*  Uc    = (float*)(ws + 200 * MB + 512 * 1024);
  float*  carry = (float*)(ws + 201 * MB);

  const int Mx = MROWS_, D = DIM_;

  // 1) precision convert: x and 4 weights to bf16
  k_cvt_bf16<<<4096, 256, 0, stream>>>(x, xb, Mx * D);
  k_cvt_bf16<<<1024, 256, 0, stream>>>(W_in,   wbin,  D * D);
  k_cvt_bf16<<<1024, 256, 0, stream>>>(W_lam,  wblam, D * D);
  k_cvt_bf16<<<1024, 256, 0, stream>>>(W_gate, wbgat, D * D);
  k_cvt_bf16<<<1024, 256, 0, stream>>>(W_out,  wbout, D * D);

  // 2) three input projections (WMMA bf16, f32 accumulate)
  dim3 gg(D / 256, Mx / 128);   // (4, 128)
  k_gemm_wmma<1, 0><<<gg, 256, 0, stream>>>(xb, wbin,  b_in,   inp,   Mx, D, D);
  k_gemm_wmma<2, 0><<<gg, 256, 0, stream>>>(xb, wblam, b_lam,  lam,   Mx, D, D);
  k_gemm_wmma<1, 1><<<gg, 256, 0, stream>>>(xb, wbgat, b_gate, gateb, Mx, D, D);

  // 3) chunked associative scan (h overwrites inp)
  dim3 gs(CHANS_ / 256, NCH_);
  k_scan_chunks <<<gs, 256, 0, stream>>>(lam, inp, Ac, Uc);
  k_scan_carries<<<CHANS_ / 256, 256, 0, stream>>>(Ac, Uc, carry);
  k_scan_apply  <<<gs, 256, 0, stream>>>(lam, inp, carry);

  // 4) LayerNorm * gate -> bf16 operand (reuse xb buffer)
  k_ln_gate<<<Mx / 4, 128, 0, stream>>>(inp, gateb, gamma, beta, xb);

  // 5) output projection -> f32 d_out
  k_gemm_wmma<0, 0><<<gg, 256, 0, stream>>>(xb, wbout, b_out, d_out, Mx, D, D);
}
